// MambaMixer2_83580063581002
// MI455X (gfx1250) — compile-verified, matrix operand order checked by reading, not run on hardware
//
#include <hip/hip_runtime.h>
#include <hip/hip_bf16.h>
#include <math.h>

// ---------- problem constants (from reference) ----------
#define H_    80
#define P_    64
#define NS_   128
#define HID_  2560
#define INTER_ 5120
#define CONV_DIM_ 5376
#define D_IN_ 10576
#define DT_OFF_ 10496           // INTER + CONV_DIM
#define BATCH_ 2
#define LSEQ_ 2048
#define MROWS_ (BATCH_*LSEQ_)   // 4096

// ---------- CDNA5 WMMA vector types ----------
typedef __attribute__((ext_vector_type(16))) __bf16 v16bf;
typedef __attribute__((ext_vector_type(8)))  __bf16 v8bf;
typedef __attribute__((ext_vector_type(8)))  float  v8f;
typedef __attribute__((ext_vector_type(4)))  unsigned int u32x4;
typedef __attribute__((ext_vector_type(8)))  unsigned int u32x8;

__device__ __forceinline__ unsigned short f32_bf16(float f) {
  unsigned int u = __float_as_uint(f);
  u += 0x7FFFu + ((u >> 16) & 1u);   // round-to-nearest-even
  return (unsigned short)(u >> 16);
}

// =====================================================================
// fp32 -> bf16 bulk convert (bandwidth bound, done ONCE per operand)
// =====================================================================
__global__ __launch_bounds__(256)
void cvt_f32_bf16(const float* __restrict__ s, unsigned short* __restrict__ d, long n) {
  const long i = ((long)blockIdx.x * 256 + threadIdx.x) * 4;
  if (i >= n) return;
  const float4 v = *(const float4*)(s + i);
  const unsigned int p0 = (unsigned int)f32_bf16(v.x) | ((unsigned int)f32_bf16(v.y) << 16);
  const unsigned int p1 = (unsigned int)f32_bf16(v.z) | ((unsigned int)f32_bf16(v.w) << 16);
  *(uint2*)(d + i) = make_uint2(p0, p1);
}

// =====================================================================
// GEMM tiling config (shared by both GEMM variants)
// Block tile 128x256x32, 256 threads = 8 waves (2x4), wave tile 64x64
// => 16 x v_wmma_f32_16x16x32_bf16 per K-chunk per wave
// =====================================================================
#define BM 128
#define BN 256
#define BK 32
#define LDSK 40   // pad 32 -> 40 shorts (80B rows) to spread LDS banks

// ---- shared fragment-compute body -------------------------
#define GEMM_COMPUTE_FRAGS()                                                     \
  do {                                                                           \
    v16bf af[4];                                                                 \
    _Pragma("unroll")                                                            \
    for (int ti = 0; ti < 4; ++ti) {                                             \
      const int r = wm * 64 + ti * 16 + l16;                                     \
      v8bf lo = *(const v8bf*)&As[r][half * 8];                                  \
      v8bf hi = *(const v8bf*)&As[r][half * 8 + 16];                             \
      af[ti] = __builtin_shufflevector(lo, hi,                                   \
               0,1,2,3,4,5,6,7,8,9,10,11,12,13,14,15);                           \
    }                                                                            \
    _Pragma("unroll")                                                            \
    for (int tj = 0; tj < 4; ++tj) {                                             \
      const int r = wn * 64 + tj * 16 + l16;                                     \
      v8bf lo = *(const v8bf*)&Bs[r][half * 8];                                  \
      v8bf hi = *(const v8bf*)&Bs[r][half * 8 + 16];                             \
      v16bf bfg = __builtin_shufflevector(lo, hi,                                \
                  0,1,2,3,4,5,6,7,8,9,10,11,12,13,14,15);                        \
      _Pragma("unroll")                                                          \
      for (int ti = 0; ti < 4; ++ti)                                             \
        acc[ti][tj] = __builtin_amdgcn_wmma_f32_16x16x32_bf16(                   \
            false, af[ti], false, bfg, (short)0, acc[ti][tj], false, false);     \
    }                                                                            \
  } while (0)

// =====================================================================
// Variant 1: vmem-staged bf16 WMMA GEMM (handles ragged N, used for
// in_proj where N=10576 is not a tile multiple)
// =====================================================================
__global__ __launch_bounds__(256)
void gemm_bf16_wmma(const unsigned short* __restrict__ A,
                    const unsigned short* __restrict__ Bw,
                    float* __restrict__ C, int M, int N, int K,
                    int lda, int ldb, int ldc) {
  __shared__ __align__(16) unsigned short As[BM][LDSK];
  __shared__ __align__(16) unsigned short Bs[BN][LDSK];

  const int tid  = threadIdx.x;
  const int lane = tid & 31;
  const int wave = tid >> 5;
  const int wm   = wave >> 2;        // 0..1  (M dir)
  const int wn   = wave & 3;         // 0..3  (N dir)
  const int half = lane >> 4;        // ISA 16-bit A layout: K half select
  const int l16  = lane & 15;
  const int m0 = blockIdx.y * BM;
  const int n0 = blockIdx.x * BN;

  const v8f vz = {0.f,0.f,0.f,0.f,0.f,0.f,0.f,0.f};
  v8f acc[4][4];
  #pragma unroll
  for (int i = 0; i < 4; ++i)
    #pragma unroll
    for (int j = 0; j < 4; ++j) acc[i][j] = vz;

  const int crow = tid >> 2;         // 0..63
  const int cq   = (tid & 3) * 8;    // short offset 0,8,16,24
  const uint4 uz = make_uint4(0u, 0u, 0u, 0u);

  for (int kb = 0; kb < K; kb += BK) {
    // ---- stage bf16 tiles into LDS: pure b128 copies ----
    #pragma unroll
    for (int pass = 0; pass < 2; ++pass) {          // A: 128 rows
      const int r = pass * 64 + crow;
      const int g = m0 + r;
      uint4 v = uz;
      if (g < M) {
        v = *(const uint4*)(A + (size_t)g * lda + kb + cq);
        if (kb + BK < K)
          __builtin_prefetch((const void*)(A + (size_t)g * lda + kb + BK + cq), 0, 1);
      }
      *(uint4*)&As[r][cq] = v;
    }
    #pragma unroll
    for (int pass = 0; pass < 4; ++pass) {          // B: 256 rows
      const int r = pass * 64 + crow;
      const int g = n0 + r;
      uint4 v = uz;
      if (g < N) {
        v = *(const uint4*)(Bw + (size_t)g * ldb + kb + cq);
        if (kb + BK < K)
          __builtin_prefetch((const void*)(Bw + (size_t)g * ldb + kb + BK + cq), 0, 1);
      }
      *(uint4*)&Bs[r][cq] = v;
    }
    __syncthreads();
    GEMM_COMPUTE_FRAGS();
    __syncthreads();
  }

  // ---- store: C/D 16x16 f32 layout: VGPR j -> (m = half*8 + j, n = lane&15)
  #pragma unroll
  for (int ti = 0; ti < 4; ++ti)
    #pragma unroll
    for (int tj = 0; tj < 4; ++tj) {
      const int col = n0 + wn * 64 + tj * 16 + l16;
      if (col < N) {
        const int rbase = m0 + wm * 64 + ti * 16 + half * 8;
        #pragma unroll
        for (int j = 0; j < 8; ++j) {
          const int row = rbase + j;
          if (row < M) C[(size_t)row * ldc + col] = acc[ti][tj][j];
        }
      }
    }
}

// =====================================================================
// Variant 2: TDM-staged GEMM (dims must be exact tile multiples).
// Wave 0 issues two tensor_load_to_lds (A tile 128x32, B tile 256x32);
// the Tensor Data Mover writes LDS directly (pad_interval=16 DW /
// pad_amount=4 DW reproduces the 80B padded row stride), bypassing the
// register file; tracked with TENSORcnt.
// =====================================================================
__device__ __forceinline__ void tdm_load_tile_2d(unsigned lds_addr,
                                                 const unsigned short* gptr,
                                                 unsigned rows, unsigned lda_elems) {
  const unsigned long long ga = (unsigned long long)(uintptr_t)(const void*)gptr;
  u32x4 g0;
  g0[0] = 1u;                                            // count=1, user mode
  g0[1] = lds_addr;                                      // LDS byte address
  g0[2] = (unsigned)ga;                                  // global_addr[31:0]
  g0[3] = (unsigned)((ga >> 32) & 0x01FFFFFFull)         // global_addr[56:32]
        | (2u << 30);                                    // type=2 ("image")
  u32x8 g1;
  g1[0] = (1u << 16)            // data_size = 1 -> 2 bytes/elem
        | (1u << 20)            // pad_enable
        | (3u << 22)            // pad_interval: 16 DWORDs (= one 64B row)
        | (3u << 25);           // pad_amount: 4 DWORDs (= 16B) -> 80B row
  g1[1] = (lda_elems & 0xFFFFu) << 16;                   // tensor_dim0[15:0]
  g1[2] = (lda_elems >> 16);                             // tensor_dim0[31:16], dim1 lo=0
  g1[3] = 16u                                            // tensor_dim1[31:16] -> 1M rows
        | (32u << 16);                                   // tile_dim0 = 32 elems
  g1[4] = rows;                                          // tile_dim1 (tile_dim2=0)
  g1[5] = lda_elems;                                     // tensor_dim0_stride[31:0]
  g1[6] = 0u;                                            // stride hi / dim1_stride lo
  g1[7] = 0u;
  asm volatile("tensor_load_to_lds %0, %1" :: "s"(g0), "s"(g1) : "memory");
}

__global__ __launch_bounds__(256)
void gemm_bf16_wmma_tdm(const unsigned short* __restrict__ A,
                        const unsigned short* __restrict__ Bw,
                        float* __restrict__ C, int M, int N, int K,
                        int lda, int ldb, int ldc) {
  __shared__ __align__(16) unsigned short As[BM][LDSK];
  __shared__ __align__(16) unsigned short Bs[BN][LDSK];

  const int tid  = threadIdx.x;
  const int lane = tid & 31;
  const int wave = tid >> 5;
  const int wm   = wave >> 2;
  const int wn   = wave & 3;
  const int half = lane >> 4;
  const int l16  = lane & 15;
  const int m0 = blockIdx.y * BM;
  const int n0 = blockIdx.x * BN;

  const v8f vz = {0.f,0.f,0.f,0.f,0.f,0.f,0.f,0.f};
  v8f acc[4][4];
  #pragma unroll
  for (int i = 0; i < 4; ++i)
    #pragma unroll
    for (int j = 0; j < 4; ++j) acc[i][j] = vz;

  const unsigned ldsA = (unsigned)(uintptr_t)(void*)&As[0][0];
  const unsigned ldsB = (unsigned)(uintptr_t)(void*)&Bs[0][0];

  for (int kb = 0; kb < K; kb += BK) {
    if (tid < 32) {                       // one wave drives the TDM
      tdm_load_tile_2d(ldsA, A  + (size_t)m0 * lda + kb, BM, (unsigned)lda);
      tdm_load_tile_2d(ldsB, Bw + (size_t)n0 * ldb + kb, BN, (unsigned)ldb);
      __builtin_amdgcn_s_wait_tensorcnt(0);
    }
    __syncthreads();
    GEMM_COMPUTE_FRAGS();
    __syncthreads();
  }

  #pragma unroll
  for (int ti = 0; ti < 4; ++ti)
    #pragma unroll
    for (int tj = 0; tj < 4; ++tj) {
      const int col = n0 + wn * 64 + tj * 16 + l16;
      const int rbase = m0 + wm * 64 + ti * 16 + half * 8;
      #pragma unroll
      for (int j = 0; j < 8; ++j)
        C[(size_t)(rbase + j) * ldc + col] = acc[ti][tj][j];
    }
}

// =====================================================================
// causal depthwise conv1d (K=4) + bias + silu over the [x,B,C] channels
// =====================================================================
__global__ __launch_bounds__(256)
void conv_silu(const float* __restrict__ proj, const float* __restrict__ cw,
               const float* __restrict__ cb, float* __restrict__ xBC) {
  const long idx = (long)blockIdx.x * 256 + threadIdx.x;
  const long total = (long)MROWS_ * CONV_DIM_;
  if (idx >= total) return;
  const int  c   = (int)(idx % CONV_DIM_);
  const long row = idx / CONV_DIM_;      // b*L + l
  const int  l   = (int)(row % LSEQ_);
  float s = cb[c];
  #pragma unroll
  for (int k = 0; k < 4; ++k) {
    const int t = l - 3 + k;
    if (t >= 0)
      s += cw[c * 4 + k] * proj[(row + (long)(t - l)) * D_IN_ + INTER_ + c];
  }
  xBC[idx] = s / (1.f + expf(-s));       // silu
}

__global__ void zero_f32(float* p, int n) {
  const int i = blockIdx.x * 256 + threadIdx.x;
  if (i < n) p[i] = 0.f;
}

// =====================================================================
// selective scan: one block per (b,h); state h[64][128] in registers
// =====================================================================
__global__ __launch_bounds__(256)
void ssm_scan(const float* __restrict__ proj, const float* __restrict__ xBC,
              const float* __restrict__ Av, const float* __restrict__ Dv,
              const float* __restrict__ dt_bias,
              float* __restrict__ y, float* __restrict__ sumsq) {
  const int b  = blockIdx.x / H_;
  const int hh = blockIdx.x % H_;
  const int tid = threadIdx.x;
  const int p  = tid >> 2;               // 0..63
  const int nc = (tid & 3) * 32;         // N chunk origin

  __shared__ float sx[P_];
  __shared__ float sB[NS_];
  __shared__ float sC[NS_];
  __shared__ float ssum;

  float st[32];
  #pragma unroll
  for (int j = 0; j < 32; ++j) st[j] = 0.f;

  const float a     = Av[hh];
  const float dskip = Dv[hh];
  const float dtb   = dt_bias[hh];

  for (int l = 0; l < LSEQ_; ++l) {
    const long row = (long)b * LSEQ_ + l;
    const float* xr = xBC + row * CONV_DIM_;
    if (tid < P_)            sx[tid]            = xr[hh * P_ + tid];
    else if (tid < P_ + NS_) sB[tid - P_]       = xr[INTER_ + (tid - P_)];
    else                     sC[tid - P_ - NS_] = xr[INTER_ + NS_ + (tid - P_ - NS_)];
    if (tid < 64)            sC[64 + tid]       = xr[INTER_ + NS_ + 64 + tid];
    if (tid == 0) ssum = 0.f;

    const float dtraw = proj[row * D_IN_ + DT_OFF_ + hh] + dtb;  // broadcast load
    const float dt = (dtraw > 20.f) ? dtraw : log1pf(expf(dtraw));
    const float dA = expf(dt * a);
    __syncthreads();

    const float xp  = sx[p];
    const float dtx = dt * xp;
    float accv = 0.f;
    #pragma unroll
    for (int j = 0; j < 32; ++j) {
      const float s2 = st[j] * dA + dtx * sB[nc + j];
      st[j] = s2;
      accv += s2 * sC[nc + j];
    }
    accv += __shfl_xor(accv, 1, 32);
    accv += __shfl_xor(accv, 2, 32);
    if ((tid & 3) == 0) {
      float yv = accv + dskip * xp;
      const float g = proj[row * D_IN_ + hh * P_ + p];   // gate
      yv *= g / (1.f + expf(-g));                        // * silu(gate)
      y[row * INTER_ + hh * P_ + p] = yv;
      atomicAdd(&ssum, yv * yv);                         // ds_add_f32
    }
    __syncthreads();
    if (tid == 0) atomicAdd(&sumsq[row], ssum);
  }
}

// =====================================================================
// gated RMSNorm scale (row-wise), emits bf16 for the out-proj GEMM
// =====================================================================
__global__ __launch_bounds__(256)
void rmsnorm_bf16(const float* __restrict__ y, const float* __restrict__ ss,
                  const float* __restrict__ nw, unsigned short* __restrict__ out) {
  const long row = blockIdx.x;
  const float scale = rsqrtf(ss[row] / (float)INTER_ + 1e-5f);
  for (int i = threadIdx.x * 2; i < INTER_; i += 512) {
    const float v0 = y[row * INTER_ + i]     * scale * nw[i];
    const float v1 = y[row * INTER_ + i + 1] * scale * nw[i + 1];
    const unsigned int pk = (unsigned int)f32_bf16(v0) | ((unsigned int)f32_bf16(v1) << 16);
    *(unsigned int*)(out + row * INTER_ + i) = pk;
  }
}

// =====================================================================
extern "C" void kernel_launch(void* const* d_in, const int* in_sizes, int n_in,
                              void* d_out, int out_size, void* d_ws, size_t ws_size,
                              hipStream_t stream) {
  const float* hs   = (const float*)d_in[0];  // [2,2048,2560]
  const float* Win  = (const float*)d_in[1];  // [10576,2560]
  const float* cw   = (const float*)d_in[2];  // [5376,4]
  const float* cb   = (const float*)d_in[3];  // [5376]
  const float* Av   = (const float*)d_in[4];  // [80] (already negative)
  const float* Dv   = (const float*)d_in[5];  // [80]
  const float* dtb  = (const float*)d_in[6];  // [80]
  const float* nw   = (const float*)d_in[7];  // [5120]
  const float* Wout = (const float*)d_in[8];  // [2560,5120]
  float* out = (float*)d_out;

  char* ws = (char*)d_ws;
  size_t off = 0;
  float* proj = (float*)(ws + off);  off += (size_t)MROWS_ * D_IN_ * sizeof(float);
  float* xBC  = (float*)(ws + off);  off += (size_t)MROWS_ * CONV_DIM_ * sizeof(float);
  float* ybuf = (float*)(ws + off);  off += (size_t)MROWS_ * INTER_ * sizeof(float);
  float* ssq  = (float*)(ws + off);  off += 4096 * sizeof(float);
  unsigned short* hsb   = (unsigned short*)(ws + off); off += (size_t)MROWS_ * HID_ * 2;
  unsigned short* Winb  = (unsigned short*)(ws + off); off += (size_t)D_IN_ * HID_ * 2;
  unsigned short* Woutb = (unsigned short*)(ws + off); off += (size_t)HID_ * INTER_ * 2;
  unsigned short* ynb   = (unsigned short*)proj;   // proj dead after scan; reuse

  dim3 blk(256);

  // 0) one-shot bf16 conversion of all GEMM operands (bandwidth bound)
  const long n_hs = (long)MROWS_ * HID_;
  const long n_wi = (long)D_IN_ * HID_;
  const long n_wo = (long)HID_ * INTER_;
  cvt_f32_bf16<<<dim3((unsigned)((n_hs / 4 + 255) / 256)), blk, 0, stream>>>(hs, hsb, n_hs);
  cvt_f32_bf16<<<dim3((unsigned)((n_wi / 4 + 255) / 256)), blk, 0, stream>>>(Win, Winb, n_wi);
  cvt_f32_bf16<<<dim3((unsigned)((n_wo / 4 + 255) / 256)), blk, 0, stream>>>(Wout, Woutb, n_wo);

  // 1) in_proj: proj = hs @ Win^T   (M=4096, N=10576 ragged -> vmem variant)
  dim3 g1((D_IN_ + BN - 1) / BN, MROWS_ / BM);
  gemm_bf16_wmma<<<g1, blk, 0, stream>>>(hsb, Winb, proj,
                                         MROWS_, D_IN_, HID_, HID_, HID_, D_IN_);

  // 2) depthwise causal conv + silu
  const long total = (long)MROWS_ * CONV_DIM_;
  conv_silu<<<dim3((unsigned)((total + 255) / 256)), blk, 0, stream>>>(proj, cw, cb, xBC);

  // 3) scan (+ gate, + sumsq accumulation)
  zero_f32<<<dim3((MROWS_ + 255) / 256), blk, 0, stream>>>(ssq, MROWS_);
  ssm_scan<<<dim3(BATCH_ * H_), blk, 0, stream>>>(proj, xBC, Av, Dv, dtb, ybuf, ssq);

  // 4) RMS norm -> bf16
  rmsnorm_bf16<<<dim3(MROWS_), blk, 0, stream>>>(ybuf, ssq, nw, ynb);

  // 5) out_proj: out = ynorm @ Wout^T  (4096x2560x5120, all tile-exact
  //    -> TDM-staged variant, LDS filled by the Tensor Data Mover)
  dim3 g2(HID_ / BN, MROWS_ / BM);
  gemm_bf16_wmma_tdm<<<g2, blk, 0, stream>>>(ynb, Woutb, out,
                                             MROWS_, HID_, INTER_, INTER_, INTER_, HID_);
}
